// MHACross_63900523430622
// MI455X (gfx1250) — matrix operand order repaired and checked by reading, not compiled, not run on hardware
//
#include <hip/hip_runtime.h>
#include <hip/hip_bf16.h>

#define DIM      1024
#define NHEADS   8
#define HEAD_DIM 128
#define BATCH    2
#define TLEN     2048
#define SLEN     3000
#define SPAD     3008                    // 94 * 32
#define QK_SCALE 0.29730177875068026f    // 128^-0.25
#define LN_ROPE  9.210340371976184f      // ln(10000)

typedef __attribute__((ext_vector_type(16))) __bf16 bf16x16;
typedef __attribute__((ext_vector_type(8)))  __bf16 bf16x8;
typedef __attribute__((ext_vector_type(8)))  float  f32x8;
typedef __attribute__((ext_vector_type(4)))  float  f32x4;

__device__ __forceinline__ f32x8 zero8() {
    f32x8 z;
#pragma unroll
    for (int i = 0; i < 8; ++i) z[i] = 0.0f;
    return z;
}

// D = A(16x32) * B(32x16) + C, bf16 in, f32 accumulate
__device__ __forceinline__ f32x8 wmma_bf16(bf16x16 a, bf16x16 b, f32x8 c) {
    return __builtin_amdgcn_wmma_f32_16x16x32_bf16(
        /*neg_a=*/false, a, /*neg_b=*/false, b,
        /*c_mod=*/(short)0, c, /*reuse_a=*/false, /*reuse_b=*/false);
}

// A-operand (M=lane&15; K: e<8 -> half*8+e, e>=8 -> 16+half*8+(e-8)); two 16B loads.
__device__ __forceinline__ bf16x16 load_a_bf16(const __bf16* row, int c0, int half) {
    bf16x8 lo = *(const bf16x8*)(row + c0 + half * 8);
    bf16x8 hi = *(const bf16x8*)(row + c0 + 16 + half * 8);
    bf16x16 a;
#pragma unroll
    for (int e = 0; e < 8; ++e) { a[e] = lo[e]; a[8 + e] = hi[e]; }
    return a;
}

// B-operand (N=lane&15; K: e<8 -> half*8+e, e>=8 -> 16+half*8+(e-8)); two 16B loads.
__device__ __forceinline__ bf16x16 load_b_bf16(const __bf16* row, int c0, int half) {
    bf16x8 lo = *(const bf16x8*)(row + c0 + half * 8);
    bf16x8 hi = *(const bf16x8*)(row + c0 + 16 + half * 8);
    bf16x16 v;
#pragma unroll
    for (int e = 0; e < 8; ++e) { v[e] = lo[e]; v[8 + e] = hi[e]; }
    return v;
}

// ---------------------------------------------------------------------------
// Kernel 0: one-shot f32 -> bf16 conversion (memory bound pre-pass).
// ---------------------------------------------------------------------------
__global__ __launch_bounds__(256) void cvt_f32_bf16(
    const float* __restrict__ src, __bf16* __restrict__ dst, int n8) {
    int i = blockIdx.x * blockDim.x + threadIdx.x;
    const int stride = gridDim.x * blockDim.x;
    for (; i < n8; i += stride) {
        const f32x4* p = (const f32x4*)(src + (size_t)i * 8);
        f32x4 a = p[0], b = p[1];
        bf16x8 o;
#pragma unroll
        for (int e = 0; e < 4; ++e) { o[e] = (__bf16)a[e]; o[4 + e] = (__bf16)b[e]; }
        *(bf16x8*)(dst + (size_t)i * 8) = o;
    }
}

// ---------------------------------------------------------------------------
// Kernel 1: q = x @ Wq^T, fused RoPE(pos=t) + scale, -> qh[B,H,T,D] bf16
// One wave: 32 t-rows x 128 cols (one head) -> acc[2][8]; B tile reused 2x.
// ---------------------------------------------------------------------------
__global__ __launch_bounds__(256) void qproj_rope(
    const __bf16* __restrict__ xb, const __bf16* __restrict__ Wqb, __bf16* __restrict__ qh) {
    const int lane = threadIdx.x & 31;
    const int col  = lane & 15;
    const int half = lane >> 4;
    const int wid  = (blockIdx.x * blockDim.x + threadIdx.x) >> 5;
    const int TT    = TLEN / 32;                 // 64
    const int ttile = wid % TT;
    const int h     = (wid / TT) % NHEADS;
    const int b     = wid / (TT * NHEADS);

    const __bf16* xrow0 = xb + ((size_t)b * TLEN + ttile * 32 + col) * DIM;
    const __bf16* xrow1 = xrow0 + (size_t)16 * DIM;

    f32x8 acc[2][8];
#pragma unroll
    for (int mt = 0; mt < 2; ++mt)
#pragma unroll
        for (int j = 0; j < 8; ++j) acc[mt][j] = zero8();

    for (int kc = 0; kc < DIM; kc += 32) {
        bf16x16 a0 = load_a_bf16(xrow0, kc, half);
        bf16x16 a1 = load_a_bf16(xrow1, kc, half);
#pragma unroll
        for (int j = 0; j < 8; ++j) {
            const __bf16* wrow = Wqb + (size_t)(h * HEAD_DIM + j * 16 + col) * DIM;
            bf16x16 bm = load_b_bf16(wrow, kc, half);
            acc[0][j] = wmma_bf16(a0, bm, acc[0][j]);
            acc[1][j] = wmma_bf16(a1, bm, acc[1][j]);
        }
    }

    __bf16* qbase = qh + (size_t)(b * NHEADS + h) * TLEN * HEAD_DIM;
#pragma unroll
    for (int mt = 0; mt < 2; ++mt) {
#pragma unroll
        for (int j = 0; j < 4; ++j) {
            const int d1 = j * 16 + col;         // < 64
            const int d2 = d1 + 64;
            const float inv = __expf(-((float)d1 * (1.0f / 64.0f)) * LN_ROPE);
#pragma unroll
            for (int r = 0; r < 8; ++r) {
                const int t = ttile * 32 + mt * 16 + half * 8 + r;
                const float ang = (float)t * inv;
                const float cs = __cosf(ang), sn = __sinf(ang);
                const float q1 = acc[mt][j][r], q2 = acc[mt][j + 4][r];
                qbase[(size_t)t * HEAD_DIM + d1] = (__bf16)((q1 * cs - q2 * sn) * QK_SCALE);
                qbase[(size_t)t * HEAD_DIM + d2] = (__bf16)((q2 * cs + q1 * sn) * QK_SCALE);
            }
        }
    }
}

// ---------------------------------------------------------------------------
// Kernel 2: kv = xmel @ Wkv^T, 32 s-rows x 128 cols per wave.
// og<8 -> K head (RoPE(pos=s)+scale -> kh[B,H,Sp,D]),
// og>=8 -> V head stored transposed -> vT[B,H,D,Sp]. Rows s>=S clamped (masked later).
// ---------------------------------------------------------------------------
__global__ __launch_bounds__(256) void kvproj_rope(
    const __bf16* __restrict__ xmb, const __bf16* __restrict__ Wkvb,
    __bf16* __restrict__ kh, __bf16* __restrict__ vT) {
    const int lane = threadIdx.x & 31;
    const int col  = lane & 15;
    const int half = lane >> 4;
    const int wid  = (blockIdx.x * blockDim.x + threadIdx.x) >> 5;
    const int ST    = SPAD / 32;                 // 94
    const int stile = wid % ST;
    const int og    = (wid / ST) % 16;           // 0..7 K heads, 8..15 V heads
    const int b     = wid / (ST * 16);

    int srow0 = stile * 32 + col;
    int srow1 = srow0 + 16;
    if (srow0 > SLEN - 1) srow0 = SLEN - 1;      // clamp OOB rows (finite garbage, masked)
    if (srow1 > SLEN - 1) srow1 = SLEN - 1;
    const __bf16* xrow0 = xmb + ((size_t)b * SLEN + srow0) * DIM;
    const __bf16* xrow1 = xmb + ((size_t)b * SLEN + srow1) * DIM;

    f32x8 acc[2][8];
#pragma unroll
    for (int mt = 0; mt < 2; ++mt)
#pragma unroll
        for (int j = 0; j < 8; ++j) acc[mt][j] = zero8();

    for (int kc = 0; kc < DIM; kc += 32) {
        bf16x16 a0 = load_a_bf16(xrow0, kc, half);
        bf16x16 a1 = load_a_bf16(xrow1, kc, half);
#pragma unroll
        for (int j = 0; j < 8; ++j) {
            const __bf16* wrow = Wkvb + (size_t)(og * HEAD_DIM + j * 16 + col) * DIM;
            bf16x16 bm = load_b_bf16(wrow, kc, half);
            acc[0][j] = wmma_bf16(a0, bm, acc[0][j]);
            acc[1][j] = wmma_bf16(a1, bm, acc[1][j]);
        }
    }

    if (og < 8) {                                // K: RoPE + scale
        const int h = og;
        __bf16* kb = kh + (size_t)(b * NHEADS + h) * SPAD * HEAD_DIM;
#pragma unroll
        for (int mt = 0; mt < 2; ++mt) {
#pragma unroll
            for (int j = 0; j < 4; ++j) {
                const int d1 = j * 16 + col;
                const int d2 = d1 + 64;
                const float inv = __expf(-((float)d1 * (1.0f / 64.0f)) * LN_ROPE);
#pragma unroll
                for (int r = 0; r < 8; ++r) {
                    const int s = stile * 32 + mt * 16 + half * 8 + r;
                    const float ang = (float)s * inv;
                    const float cs = __cosf(ang), sn = __sinf(ang);
                    const float k1 = acc[mt][j][r], k2 = acc[mt][j + 4][r];
                    kb[(size_t)s * HEAD_DIM + d1] = (__bf16)((k1 * cs - k2 * sn) * QK_SCALE);
                    kb[(size_t)s * HEAD_DIM + d2] = (__bf16)((k2 * cs + k1 * sn) * QK_SCALE);
                }
            }
        }
    } else {                                     // V: store transposed [D, Sp]
        const int h = og - 8;
        __bf16* vb = vT + (size_t)(b * NHEADS + h) * HEAD_DIM * SPAD;
#pragma unroll
        for (int mt = 0; mt < 2; ++mt) {
#pragma unroll
            for (int j = 0; j < 8; ++j) {
                const int d = j * 16 + col;
#pragma unroll
                for (int r = 0; r < 8; ++r) {
                    const int s = stile * 32 + mt * 16 + half * 8 + r;
                    vb[(size_t)d * SPAD + s] = (__bf16)acc[mt][j][r];
                }
            }
        }
    }
}

// ---------------------------------------------------------------------------
// Kernel 3: flash attention. One wave: 16 q-rows x D=128 output, streaming S in
// 32-wide steps (2 score tiles -> online softmax -> P through LDS -> 8 PV WMMA).
// ---------------------------------------------------------------------------
__global__ __launch_bounds__(256) void attn_flash(
    const __bf16* __restrict__ qh, const __bf16* __restrict__ kh,
    const __bf16* __restrict__ vT, __bf16* __restrict__ oh) {
    __shared__ __bf16 plds[8][16 * 32];          // 1KB P tile per wave
    const int lane  = threadIdx.x & 31;
    const int col   = lane & 15;
    const int half  = lane >> 4;
    const int wslot = threadIdx.x >> 5;
    __bf16* P = &plds[wslot][0];

    const int wid   = (blockIdx.x * blockDim.x + threadIdx.x) >> 5;
    const int TT    = TLEN / 16;
    const int ttile = wid % TT;
    const int h     = (wid / TT) % NHEADS;
    const int b     = wid / (TT * NHEADS);

    const __bf16* qbase = qh + (size_t)(b * NHEADS + h) * TLEN * HEAD_DIM;
    const __bf16* kbase = kh + (size_t)(b * NHEADS + h) * SPAD * HEAD_DIM;
    const __bf16* vbase = vT + (size_t)(b * NHEADS + h) * HEAD_DIM * SPAD;

    const int trow = ttile * 16 + col;
    bf16x16 aq[4];
#pragma unroll
    for (int kc = 0; kc < 4; ++kc)
        aq[kc] = load_a_bf16(qbase + (size_t)trow * HEAD_DIM, kc * 32, half);

    f32x8 acc[8];
    float m[8], l[8];
#pragma unroll
    for (int j = 0; j < 8; ++j) acc[j] = zero8();
#pragma unroll
    for (int r = 0; r < 8; ++r) { m[r] = -1e30f; l[r] = 0.0f; }

    for (int it = 0; it < SPAD / 32; ++it) {     // 94 iterations
        const int sbase = it * 32;
        f32x8 sc[2];
#pragma unroll
        for (int u = 0; u < 2; ++u) {
            sc[u] = zero8();
            const int srow = sbase + u * 16 + col;
            const __bf16* krow = kbase + (size_t)srow * HEAD_DIM;
#pragma unroll
            for (int kc = 0; kc < 4; ++kc)
                sc[u] = wmma_bf16(aq[kc], load_b_bf16(krow, kc * 32, half), sc[u]);
            const bool oob = srow >= SLEN;       // mask padded columns
#pragma unroll
            for (int r = 0; r < 8; ++r) sc[u][r] = oob ? -1e30f : sc[u][r];
        }

        // row-wise online softmax (row stats live replicated in each 16-lane group)
        float tm[8], p0[8], p1[8], rs[8];
#pragma unroll
        for (int r = 0; r < 8; ++r) tm[r] = fmaxf(sc[0][r], sc[1][r]);
#pragma unroll
        for (int off = 1; off < 16; off <<= 1)
#pragma unroll
            for (int r = 0; r < 8; ++r) tm[r] = fmaxf(tm[r], __shfl_xor(tm[r], off, 32));
#pragma unroll
        for (int r = 0; r < 8; ++r) {
            const float mn = fmaxf(m[r], tm[r]);
            const float cf = __expf(m[r] - mn);
            m[r] = mn;
            p0[r] = __expf(sc[0][r] - mn);
            p1[r] = __expf(sc[1][r] - mn);
            rs[r] = p0[r] + p1[r];
            l[r] *= cf;
#pragma unroll
            for (int j = 0; j < 8; ++j) acc[j][r] *= cf;
        }
#pragma unroll
        for (int off = 1; off < 16; off <<= 1)
#pragma unroll
            for (int r = 0; r < 8; ++r) rs[r] += __shfl_xor(rs[r], off, 32);
#pragma unroll
        for (int r = 0; r < 8; ++r) l[r] += rs[r];

        // P (C-layout) -> LDS row-major [16][32] -> reload in A-layout
#pragma unroll
        for (int r = 0; r < 8; ++r) {
            P[(half * 8 + r) * 32 + col]      = (__bf16)p0[r];
            P[(half * 8 + r) * 32 + 16 + col] = (__bf16)p1[r];
        }
        asm volatile("s_wait_dscnt 0" ::: "memory");
        bf16x16 ap = load_a_bf16(P + col * 32, 0, half);

#pragma unroll
        for (int j = 0; j < 8; ++j) {
            const __bf16* vrow = vbase + (size_t)(j * 16 + col) * SPAD + sbase;
            acc[j] = wmma_bf16(ap, load_b_bf16(vrow, 0, half), acc[j]);
        }
    }

    float rl[8];
#pragma unroll
    for (int r = 0; r < 8; ++r) rl[r] = 1.0f / l[r];

    // oh layout [B,T,H,D] == [B,T,C] contiguous
#pragma unroll
    for (int j = 0; j < 8; ++j) {
        const int d = j * 16 + col;
#pragma unroll
        for (int r = 0; r < 8; ++r) {
            const int t = ttile * 16 + half * 8 + r;
            oh[((size_t)(b * TLEN + t) * NHEADS + h) * HEAD_DIM + d] = (__bf16)(acc[j][r] * rl[r]);
        }
    }
}

// ---------------------------------------------------------------------------
// Kernel 4: y = oh @ Wout^T  (bf16 A/B, f32 out), 32 rows x 128 cols per wave.
// ---------------------------------------------------------------------------
__global__ __launch_bounds__(256) void outproj(
    const __bf16* __restrict__ oh, const __bf16* __restrict__ Woutb, float* __restrict__ y) {
    const int lane = threadIdx.x & 31;
    const int col  = lane & 15;
    const int half = lane >> 4;
    const int wid  = (blockIdx.x * blockDim.x + threadIdx.x) >> 5;
    const int TT    = TLEN / 32;                 // 64
    const int g     = wid % 8;                   // 128-col block
    const int ttile = (wid / 8) % TT;
    const int b     = wid / (8 * TT);

    const __bf16* arow0 = oh + (size_t)(b * TLEN + ttile * 32 + col) * DIM;
    const __bf16* arow1 = arow0 + (size_t)16 * DIM;

    f32x8 acc[2][8];
#pragma unroll
    for (int mt = 0; mt < 2; ++mt)
#pragma unroll
        for (int j = 0; j < 8; ++j) acc[mt][j] = zero8();

    for (int kc = 0; kc < DIM; kc += 32) {
        bf16x16 a0 = load_a_bf16(arow0, kc, half);
        bf16x16 a1 = load_a_bf16(arow1, kc, half);
#pragma unroll
        for (int j = 0; j < 8; ++j) {
            const __bf16* wrow = Woutb + (size_t)(g * HEAD_DIM + j * 16 + col) * DIM;
            bf16x16 bm = load_b_bf16(wrow, kc, half);
            acc[0][j] = wmma_bf16(a0, bm, acc[0][j]);
            acc[1][j] = wmma_bf16(a1, bm, acc[1][j]);
        }
    }

#pragma unroll
    for (int mt = 0; mt < 2; ++mt) {
#pragma unroll
        for (int j = 0; j < 8; ++j) {
            const int i = g * HEAD_DIM + j * 16 + col;
#pragma unroll
            for (int r = 0; r < 8; ++r) {
                const int t = ttile * 32 + mt * 16 + half * 8 + r;
                y[(size_t)(b * TLEN + t) * DIM + i] = acc[mt][j][r];
            }
        }
    }
}

extern "C" void kernel_launch(void* const* d_in, const int* in_sizes, int n_in,
                              void* d_out, int out_size, void* d_ws, size_t ws_size,
                              hipStream_t stream) {
    const float* x    = (const float*)d_in[0];
    const float* xmel = (const float*)d_in[1];
    const float* Wq   = (const float*)d_in[2];
    const float* Wkv  = (const float*)d_in[3];
    const float* Wout = (const float*)d_in[4];
    float* y = (float*)d_out;

    const size_t qh_e  = (size_t)BATCH * NHEADS * TLEN * HEAD_DIM;  // 4.19M
    const size_t kh_e  = (size_t)BATCH * NHEADS * SPAD * HEAD_DIM;  // 6.16M
    const size_t x_e   = (size_t)BATCH * TLEN * DIM;                // 4.19M
    const size_t xm_e  = (size_t)BATCH * SLEN * DIM;                // 6.14M
    const size_t wq_e  = (size_t)DIM * DIM;                         // 1.05M
    const size_t wkv_e = (size_t)2 * DIM * DIM;                     // 2.10M

    char* ws = (char*)d_ws;
    __bf16* qh   = (__bf16*)ws;  ws += qh_e  * sizeof(__bf16);
    __bf16* kh   = (__bf16*)ws;  ws += kh_e  * sizeof(__bf16);
    __bf16* vT   = (__bf16*)ws;  ws += kh_e  * sizeof(__bf16);
    __bf16* oh   = (__bf16*)ws;  ws += qh_e  * sizeof(__bf16);
    __bf16* xb   = (__bf16*)ws;  ws += x_e   * sizeof(__bf16);
    __bf16* xmb  = (__bf16*)ws;  ws += xm_e  * sizeof(__bf16);
    __bf16* wqb  = (__bf16*)ws;  ws += wq_e  * sizeof(__bf16);
    __bf16* wkvb = (__bf16*)ws;  ws += wkv_e * sizeof(__bf16);
    __bf16* wob  = (__bf16*)ws;

    const dim3 blk(256);
    // one-shot bf16 conversion pre-passes (memory bound, ~87MB total traffic)
    cvt_f32_bf16<<<1024, blk, 0, stream>>>(x,    xb,   (int)(x_e   / 8));
    cvt_f32_bf16<<<1024, blk, 0, stream>>>(xmel, xmb,  (int)(xm_e  / 8));
    cvt_f32_bf16<<<512,  blk, 0, stream>>>(Wq,   wqb,  (int)(wq_e  / 8));
    cvt_f32_bf16<<<512,  blk, 0, stream>>>(Wkv,  wkvb, (int)(wkv_e / 8));
    cvt_f32_bf16<<<512,  blk, 0, stream>>>(Wout, wob,  (int)(wq_e  / 8));

    qproj_rope <<<(BATCH * NHEADS * (TLEN / 32)) / 8, blk, 0, stream>>>(xb, wqb, qh);
    kvproj_rope<<<(BATCH * 16 * (SPAD / 32)) / 8,     blk, 0, stream>>>(xmb, wkvb, kh, vT);
    attn_flash <<<(BATCH * NHEADS * (TLEN / 16)) / 8, blk, 0, stream>>>(qh, kh, vT, oh);
    outproj    <<<(BATCH * (TLEN / 32) * 8) / 8,      blk, 0, stream>>>(oh, wob, y);
}
